// CBMGCNInference_5875515261387
// MI455X (gfx1250) — matrix operand order repaired and checked by reading, not compiled, not run on hardware
//
#include <hip/hip_runtime.h>

#define N_NODES 100000
#define N_EDGES 3200000
#define IN_F 128
#define OUT_F 64
#define M_TILES (N_NODES / 16) // 6250, exact

typedef __attribute__((ext_vector_type(16))) __bf16 v16bf;
typedef __attribute__((ext_vector_type(8)))  float  v8f;

// ---------------------------------------------------------------- utilities
__global__ void zero_f32(float* __restrict__ p, int n) {
  int i = blockIdx.x * blockDim.x + threadIdx.x;
  if (i < n) p[i] = 0.0f;
}

// ------------------------------------------------------------- degree count
__global__ void degree_kernel(const long long* __restrict__ dst,
                              float* __restrict__ deg, int n_edges) {
  int e = blockIdx.x * blockDim.x + threadIdx.x;
  if (e < n_edges) {
    unsafeAtomicAdd(&deg[(int)dst[e]], 1.0f); // global_atomic_add_f32
  }
}

__global__ void dinv_kernel(const float* __restrict__ deg,
                            float* __restrict__ dinv, int n) {
  int i = blockIdx.x * blockDim.x + threadIdx.x;
  if (i < n) {
    float d = deg[i];
    dinv[i] = (d > 0.0f) ? rsqrtf(d) : 0.0f;
  }
}

// ---------------------------------------------------- GEMM: h = x @ W^T
// Split-precision bf16 WMMA: x = hi + lo (bf16 each); keep hi*hi + hi*lo +
// lo*hi for ~fp32 accuracy on the bf16 matrix pipe.
// One wave computes a 16(M) x 64(N) tile of h. K = 128 = 4 chunks of 32.
__global__ __launch_bounds__(256) void gemm_wmma(const float* __restrict__ x,
                                                 const float* __restrict__ W,
                                                 float* __restrict__ h) {
  __shared__ __align__(64) __bf16 whi[OUT_F * IN_F]; // 16 KB
  __shared__ __align__(64) __bf16 wlo[OUT_F * IN_F]; // 16 KB

  // Stage W into LDS, pre-split into bf16 hi/lo (W^T tile source).
  for (int i = threadIdx.x; i < OUT_F * IN_F; i += 256) {
    float w = W[i];
    __bf16 hb = (__bf16)w;
    whi[i] = hb;
    wlo[i] = (__bf16)(w - (float)hb);
  }
  __syncthreads();

  const int lane   = threadIdx.x & 31;
  const int wave   = threadIdx.x >> 5;
  const int tile   = blockIdx.x * 8 + wave;
  const int tile_c = (tile < M_TILES) ? tile : (M_TILES - 1); // clamp: keep EXEC all-1s
  const int m0     = tile_c * 16;
  const bool hiHalf = (lane >= 16);
  const int  mlane  = lane & 15;

  // A-matrix row this lane feeds (both half-waves cover M = m0..m0+15).
  const float4* xr = (const float4*)(x + (size_t)(m0 + mlane) * IN_F);

  v8f acc[4] = {}; // four 16x16 f32 accumulators (N tiles 0..3)

#pragma unroll
  for (int kc = 0; kc < 4; ++kc) {
    // ---- A tile 16x32 bf16, contiguous k order (same as B):
    // lanes 0-15 : K = kc*32 + 0..15
    // lanes 16-31: K = kc*32 + 16..31
    const int kbA = kc * 32 + (hiHalf ? 16 : 0);
    float4 f0 = xr[(kbA >> 2)];
    float4 f1 = xr[(kbA >> 2) + 1];
    float4 f2 = xr[((kbA + 8) >> 2)];
    float4 f3 = xr[((kbA + 8) >> 2) + 1];
    float fa[16] = {f0.x, f0.y, f0.z, f0.w, f1.x, f1.y, f1.z, f1.w,
                    f2.x, f2.y, f2.z, f2.w, f3.x, f3.y, f3.z, f3.w};
    v16bf a_hi, a_lo;
#pragma unroll
    for (int e = 0; e < 16; ++e) {
      __bf16 hb = (__bf16)fa[e];
      a_hi[e] = hb;
      a_lo[e] = (__bf16)(fa[e] - (float)hb);
    }

    // ---- B tiles 32x16 bf16 (B[k][o] = W[o][k]), contiguous k order (same as A):
    // lanes 0-15 : N = lane,     K = kc*32 + 0..15
    // lanes 16-31: N = lane-16,  K = kc*32 + 16..31
    const int kbB = kc * 32 + (hiHalf ? 16 : 0);
#pragma unroll
    for (int n = 0; n < 4; ++n) {
      const int o = n * 16 + mlane;
      const v16bf b_hi = *(const v16bf*)&whi[o * IN_F + kbB];
      const v16bf b_lo = *(const v16bf*)&wlo[o * IN_F + kbB];
      acc[n] = __builtin_amdgcn_wmma_f32_16x16x32_bf16(
          false, a_hi, false, b_hi, (short)0, acc[n], false, false);
      acc[n] = __builtin_amdgcn_wmma_f32_16x16x32_bf16(
          false, a_hi, false, b_lo, (short)0, acc[n], false, false);
      acc[n] = __builtin_amdgcn_wmma_f32_16x16x32_bf16(
          false, a_lo, false, b_hi, (short)0, acc[n], false, false);
    }
  }

  // ---- store D (f32 16x16 layout: VGPR r -> M = r (+8 upper half), N = lane&15)
  if (tile < M_TILES) {
#pragma unroll
    for (int n = 0; n < 4; ++n) {
      const int col = n * 16 + mlane;
#pragma unroll
      for (int r = 0; r < 8; ++r) {
        const int m = m0 + r + (hiHalf ? 8 : 0);
        h[(size_t)m * OUT_F + col] = acc[n][r];
      }
    }
  }
}

// ------------------------------------------------ SpMM: y += A_norm @ h
// One wave per edge; each lane handles a float2 of the 64 features.
// h (25.6 MB) and y (25.6 MB) are L2-resident on MI455X (192 MB L2), so the
// random gather + f32 atomics resolve on-chip.
__global__ __launch_bounds__(256) void spmm_kernel(
    const long long* __restrict__ src, const long long* __restrict__ dst,
    const float* __restrict__ h, const float* __restrict__ dinv,
    float* __restrict__ y, int n_edges) {
  int t = blockIdx.x * 256 + threadIdx.x;
  int e = t >> 5;
  int lane = t & 31;
  if (e >= n_edges) return;

  // stream the edge list ahead (global_prefetch_b8, speculative)
  __builtin_prefetch(&src[e + 2048], 0, 0);
  __builtin_prefetch(&dst[e + 2048], 0, 0);

  int s = (int)src[e];
  int d = (int)dst[e];
  float norm = dinv[s] * dinv[d];

  float2 v = ((const float2*)(h + (size_t)s * OUT_F))[lane]; // coalesced 256B/edge
  float* yr = y + (size_t)d * OUT_F;
  unsafeAtomicAdd(&yr[2 * lane],     v.x * norm);
  unsafeAtomicAdd(&yr[2 * lane + 1], v.y * norm);
}

// --------------------------------------------------------------- launcher
extern "C" void kernel_launch(void* const* d_in, const int* in_sizes, int n_in,
                              void* d_out, int out_size, void* d_ws, size_t ws_size,
                              hipStream_t stream) {
  const float*     x          = (const float*)d_in[0];
  const long long* edge_index = (const long long*)d_in[1]; // int64 [2, E]
  const float*     W          = (const float*)d_in[2];
  float*           y          = (float*)d_out;

  const long long* src = edge_index;
  const long long* dst = edge_index + N_EDGES;

  // workspace layout: h [N*64] | deg [N] | dinv [N]  (~26.4 MB)
  float* h    = (float*)d_ws;
  float* deg  = h + (size_t)N_NODES * OUT_F;
  float* dinv = deg + N_NODES;

  zero_f32<<<(N_NODES * OUT_F + 255) / 256, 256, 0, stream>>>(y, N_NODES * OUT_F);
  zero_f32<<<(N_NODES + 255) / 256, 256, 0, stream>>>(deg, N_NODES);

  degree_kernel<<<(N_EDGES + 255) / 256, 256, 0, stream>>>(dst, deg, N_EDGES);
  dinv_kernel<<<(N_NODES + 255) / 256, 256, 0, stream>>>(deg, dinv, N_NODES);

  gemm_wmma<<<(M_TILES + 7) / 8, 256, 0, stream>>>(x, W, h);

  long long total = (long long)N_EDGES * 32;
  spmm_kernel<<<(int)((total + 255) / 256), 256, 0, stream>>>(src, dst, h, dinv,
                                                              y, N_EDGES);
}
